// KANLinear_609885356724
// MI455X (gfx1250) — matrix operand order repaired and checked by reading, not compiled, not run on hardware
//
#include <hip/hip_runtime.h>

// KAN linear layer as one f16 WMMA GEMM on gfx1250 (MI455X).
//
// y[b,o] = sum_i scale_base[o,i]*(x+sin x)[b,i]
//        + sum_{i,k} coeff[o,i,k]*scale_spline[o,i]*spline[b,i,k] + bias[o]
//
// S(16384 x 6144 f16, built on the fly) * W^T(512 x 6144 f16, prepacked in
// d_ws) with f32 accumulation; K-per-input = 12 (11 quartic B-spline bases +
// 1 base-activation slot). Spline bases are evaluated sparsely (only the
// (p+1)-wide nonzero Cox-de-Boor window). W tiles are DMA'd into LDS by the
// Tensor Data Mover, overlapping with spline VALU work.  B fragments are
// loaded in batches pinned by sched_barrier so the waitcnt ladder overlaps
// ds_load latency with back-to-back WMMAs.

typedef _Float16 v8h  __attribute__((ext_vector_type(8)));
typedef _Float16 v16h __attribute__((ext_vector_type(16)));
typedef float    v8f  __attribute__((ext_vector_type(8)));
typedef unsigned int u32x4 __attribute__((ext_vector_type(4)));
typedef int      i32x8 __attribute__((ext_vector_type(8)));
typedef int      i32x4 __attribute__((ext_vector_type(4)));

constexpr int IN_DIM  = 512;
constexpr int OUT_DIM = 512;
constexpr int BATCH   = 16384;
constexpr int NKNOTS  = 16;
constexpr int NBASIS  = 11;
constexpr int KPI     = 12;              // 11 spline + 1 base slot
constexpr int CHUNK_I = 8;               // input dims per K-chunk
constexpr int CK      = KPI * CHUNK_I;   // 96 = 3 WMMA k-steps
constexpr int KTOT    = IN_DIM * KPI;    // 6144
constexpr int TILE_B  = 64;
constexpr int TILE_O  = 256;
constexpr int NTHREADS = 256;            // 8 wave32s

#if defined(__HIP_DEVICE_COMPILE__) && __has_builtin(__builtin_amdgcn_tensor_load_to_lds)
#define HAVE_TDM 1
#else
#define HAVE_TDM 0
#endif

__device__ __forceinline__ float fastrcp(float x) {
#if __has_builtin(__builtin_amdgcn_rcpf)
    return __builtin_amdgcn_rcpf(x);     // v_rcp_f32
#else
    return 1.0f / x;
#endif
}

__device__ __forceinline__ void wait_tensorcnt0() {
#if __has_builtin(__builtin_amdgcn_s_wait_tensorcnt)
    __builtin_amdgcn_s_wait_tensorcnt(0);
#else
    asm volatile("s_wait_tensorcnt 0x0" ::: "memory");
#endif
}

__device__ __forceinline__ void sched_fence() {
#if __has_builtin(__builtin_amdgcn_sched_barrier)
    __builtin_amdgcn_sched_barrier(0);   // pin load-group vs wmma-group order
#endif
}

#if HAVE_TDM
// TDM load of a 2D W tile: TILE_O rows x CK f16 elems, row stride KTOT,
// into LDS at ldsAddr.  D# layout per CDNA5 ISA ch.8.  6-arg builtin:
//   (uint32x4 g0, int32x8 g1, int32x4 g2, int32x4 g3, int32x8 extra, i32 cpol)
__device__ __forceinline__ void tdm_load_w_tile(const _Float16* gsrc, unsigned ldsAddr) {
    unsigned long long ga = (unsigned long long)(__UINTPTR_TYPE__)gsrc;
    u32x4 g0 = { 1u,                                   // count=1, gather off
                 ldsAddr,                              // lds_addr [63:32]
                 (unsigned)(ga & 0xFFFFFFFFu),         // global_addr lo
                 (unsigned)((ga >> 32) & 0x1FFFFFFu) | (2u << 30) }; // +type=2
    i32x8 g1 = { (int)(1u << 16),                      // data_size = 2B
                 (int)((unsigned)(KTOT & 0xFFFF) << 16),                     // tensor_dim0 lo16
                 (int)(((KTOT >> 16) & 0xFFFF) | ((unsigned)OUT_DIM << 16)), // dim0 hi | dim1 lo
                 (int)(((OUT_DIM >> 16) & 0xFFFF) | ((unsigned)CK << 16)),   // dim1 hi | tile_dim0
                 (int)(TILE_O & 0xFFFF),               // tile_dim1 (tile_dim2=0)
                 (int)KTOT,                            // tensor_dim0_stride lo32
                 0, 0 };
    i32x4 z4 = { 0, 0, 0, 0 };
    i32x8 z8 = { 0, 0, 0, 0, 0, 0, 0, 0 };
    __builtin_amdgcn_tensor_load_to_lds(g0, g1, z4, z4, z8, 0);
}
#endif

// ---------------- pass 1: prepack W to f16 ----------------
__global__ __launch_bounds__(256)
void kan_pack_w(const float* __restrict__ coeff, const float* __restrict__ sb,
                const float* __restrict__ ss, _Float16* __restrict__ w) {
    int idx = blockIdx.x * 256 + threadIdx.x;
    if (idx >= OUT_DIM * KTOT) return;
    int o   = idx / KTOT;
    int rem = idx - o * KTOT;
    int i   = rem / KPI;
    int k   = rem - i * KPI;
    size_t oi = (size_t)o * IN_DIM + i;
    float v = (k < NBASIS) ? coeff[oi * NBASIS + k] * ss[oi] : sb[oi];
    w[idx] = (_Float16)v;
}

// ---------------- pass 2: fused spline + GEMM ----------------
template <bool PRE>
__global__ __launch_bounds__(NTHREADS)
void kan_gemm(const float* __restrict__ x, const float* __restrict__ grid,
              const float* __restrict__ coeff, const float* __restrict__ scale_base,
              const float* __restrict__ scale_spline, const float* __restrict__ bias,
              const _Float16* __restrict__ wpack, float* __restrict__ out) {
    __shared__ _Float16 Slds[TILE_B * CK];        // 12 KB activations (A)
    __shared__ _Float16 Wlds[TILE_O * CK];        // 48 KB weights (B)
    __shared__ float    Gs[CHUNK_I][NKNOTS];      // knots
    __shared__ float    InvD[CHUNK_I][4][15];     // 1/(g[j+p]-g[j]), 0 if diff==0

    const int tid     = threadIdx.x;
    const int lane    = tid & 31;
    const int wave    = tid >> 5;
    const int rowWave = wave & 3;                 // 4 M-tiles of 16
    const int colWave = wave >> 2;                // 2 halves of 128 cols
    const int bBase   = blockIdx.y * TILE_B;
    const int oBase   = blockIdx.x * TILE_O;

    const int lan15 = lane & 15;
    const int lhi   = lane >> 4;
    // fragment base pointers (constant across chunks)
    const _Float16* paBase = &Slds[(rowWave * 16 + lan15) * CK + lhi * 8];
    const _Float16* pbBase = &Wlds[(colWave * 128 + lan15) * CK + lhi * 8];

    v8f acc[8];
    {
        v8f z = {};
        #pragma unroll
        for (int tn = 0; tn < 8; ++tn) acc[tn] = z;
    }

    constexpr int nChunks = IN_DIM / CHUNK_I;     // 64
    for (int ch = 0; ch < nChunks; ++ch) {
        const int iBase = ch * CHUNK_I;
        __syncthreads();                           // LDS readers of prev chunk done

        if constexpr (PRE) {
#if HAVE_TDM
            if (wave == 0) {                       // one TDM issue per workgroup
                const _Float16* gsrc = wpack + (size_t)oBase * KTOT + (size_t)iBase * KPI;
                unsigned ldsAddr = (unsigned)(__UINTPTR_TYPE__)(void*)Wlds; // low 32b = LDS offset
                tdm_load_w_tile(gsrc, ldsAddr);
            }
#else
            // plain copy fallback: 64-bit chunks, 24 per thread
            const _Float16* wsrc = wpack + (size_t)oBase * KTOT + (size_t)iBase * KPI;
            #pragma unroll 4
            for (int t = 0; t < (TILE_O * CK) / (NTHREADS * 4); ++t) {
                int c   = tid + t * NTHREADS;      // 64-bit chunk id
                int row = c / (CK / 4);
                int off = (c - row * (CK / 4)) * 4;
                *(unsigned long long*)&Wlds[row * CK + off] =
                    *(const unsigned long long*)&wsrc[(size_t)row * KTOT + off];
            }
#endif
        } else {
            // fused f32->f16 conversion of the W tile
            #pragma unroll 4
            for (int t = 0; t < (TILE_O * CK) / NTHREADS; ++t) {
                int e   = tid + t * NTHREADS;
                int oL  = e / CK;
                int rem = e - oL * CK;
                int iL  = rem / KPI;
                int k   = rem - iL * KPI;
                size_t oi = (size_t)(oBase + oL) * IN_DIM + (iBase + iL);
                float v = (k < NBASIS) ? coeff[oi * NBASIS + k] * scale_spline[oi]
                                       : scale_base[oi];
                Wlds[oL * CK + rem] = (_Float16)v;
            }
        }

        // knot tables: one rcp per node, batch-invariant
        if (tid < CHUNK_I) {
            const float* gr = grid + (size_t)(iBase + tid) * NKNOTS;
            float gg[NKNOTS];
            #pragma unroll
            for (int j = 0; j < NKNOTS; ++j) { gg[j] = gr[j]; Gs[tid][j] = gg[j]; }
            #pragma unroll
            for (int p = 1; p <= 4; ++p)
                for (int j = 0; j <= NKNOTS - 1 - p; ++j) {
                    float d = gg[j + p] - gg[j];
                    InvD[tid][p - 1][j] = (d != 0.0f) ? fastrcp(d) : 0.0f;
                }
        }
        __syncthreads();                           // tables visible

        // ---- sparse Cox-de-Boor: 2 evals/thread, only the nonzero window ----
        #pragma unroll
        for (int t = 0; t < (TILE_B * CHUNK_I) / NTHREADS; ++t) {
            int pr = tid + t * NTHREADS;
            int bL = pr >> 3;
            int iL = pr & 7;
            float xv = x[(size_t)(bBase + bL) * IN_DIM + iBase + iL];
            // speculative prefetch of next chunk's x (dropped if invalid)
            __builtin_prefetch(&x[(size_t)(bBase + bL) * IN_DIM + iBase + CHUNK_I + iL], 0, 0);
            const float* gk = &Gs[iL][0];
            const float* iv = &InvD[iL][0][0];
            int j0 = -1;                           // hot interval: g[j0] <= x < g[j0+1]
            #pragma unroll
            for (int j = 0; j < NKNOTS; ++j) j0 += (xv >= gk[j]) ? 1 : 0;
            float Wc[5];
            Wc[0] = (j0 >= 0 && j0 <= NKNOTS - 2) ? 1.0f : 0.0f;
            Wc[1] = Wc[2] = Wc[3] = Wc[4] = 0.0f;
            #pragma unroll
            for (int p = 1; p <= 4; ++p) {
                float Wn[5];
                #pragma unroll
                for (int d = 0; d <= p; ++d) {
                    int  j  = j0 - p + d;
                    bool ok = (j >= 0) && (j <= (NKNOTS - 2) - p);
                    int  jc = ok ? j : 0;
                    float vl = (d >= 1)     ? Wc[d - 1] : 0.0f;
                    float vr = (d <= p - 1) ? Wc[d]     : 0.0f;
                    float rl = (xv - gk[jc])         * iv[(p - 1) * 15 + jc];
                    float rr = (gk[jc + p + 1] - xv) * iv[(p - 1) * 15 + jc + 1];
                    float val = (rl * vl + rr * vr) * fastrcp(rl + rr + 1e-8f);
                    Wn[d] = ok ? val : 0.0f;
                }
                #pragma unroll
                for (int d = 0; d <= p; ++d) Wc[d] = Wn[d];
            }
            _Float16* sp = &Slds[bL * CK + iL * KPI];
            *(unsigned long long*)(sp)     = 0ull;         // bases 0..3
            *(unsigned long long*)(sp + 4) = 0ull;         // bases 4..7
            *(unsigned int*)(sp + 8)       = 0u;           // bases 8..9
            sp[10] = (_Float16)0.0f;
            #pragma unroll
            for (int d = 0; d < 5; ++d) {                  // scatter window
                int j = j0 - 4 + d;
                if (j >= 0 && j <= NBASIS - 1) sp[j] = (_Float16)Wc[d];
            }
            sp[NBASIS] = (_Float16)(xv + __sinf(xv));      // base activation
        }

#if HAVE_TDM
        if constexpr (PRE) { if (wave == 0) wait_tensorcnt0(); }
#endif
        __syncthreads();                           // S + W tiles ready

        // ---- WMMA: 3 k-steps; per k-step load all 8 B fragments as a batch
        //      (pinned by sched_barrier) then issue 8 WMMAs back-to-back.
        //      LDS loads retire in order -> waitcnt ladder, not full drains. ----
        v16h a3[3];
        #pragma unroll
        for (int ks = 0; ks < 3; ++ks) {
            const _Float16* pa = paBase + ks * 32;
            v8h alo = *(const v8h*)(pa);
            v8h ahi = *(const v8h*)(pa + 16);
            a3[ks] = __builtin_shufflevector(alo, ahi,
                     0,1,2,3,4,5,6,7,8,9,10,11,12,13,14,15);
        }
        #pragma unroll
        for (int ks = 0; ks < 3; ++ks) {
            v16h b[8];
            #pragma unroll
            for (int tn = 0; tn < 8; ++tn) {
                const _Float16* pb = pbBase + tn * 16 * CK + ks * 32;
                v8h blo = *(const v8h*)(pb);
                v8h bhi = *(const v8h*)(pb + 16);
                b[tn] = __builtin_shufflevector(blo, bhi,
                        0,1,2,3,4,5,6,7,8,9,10,11,12,13,14,15);
            }
            sched_fence();   // keep the 16 ds_loads ahead of the WMMA block
            #pragma unroll
            for (int tn = 0; tn < 8; ++tn)
                acc[tn] = __builtin_amdgcn_wmma_f32_16x16x32_f16(
                    false, a3[ks], false, b[tn], (short)0, acc[tn], false, false);
            sched_fence();
        }
    }

    // ---- epilogue: C f32 16x16: VGPR r -> M=r+8*(lane>=16), N=lane&15 ----
    int bRow0 = bBase + rowWave * 16 + (lhi * 8);
    #pragma unroll
    for (int tn = 0; tn < 8; ++tn) {
        int o = oBase + colWave * 128 + tn * 16 + lan15;
        float bv = bias[o];
        #pragma unroll
        for (int r = 0; r < 8; ++r)
            out[(size_t)(bRow0 + r) * OUT_DIM + o] = acc[tn][r] + bv;
    }
}

extern "C" void kernel_launch(void* const* d_in, const int* in_sizes, int n_in,
                              void* d_out, int out_size, void* d_ws, size_t ws_size,
                              hipStream_t stream) {
    (void)in_sizes; (void)n_in; (void)out_size;
    const float* x  = (const float*)d_in[0];
    const float* g  = (const float*)d_in[1];
    const float* cf = (const float*)d_in[2];
    const float* sb = (const float*)d_in[3];
    const float* ss = (const float*)d_in[4];
    const float* bs = (const float*)d_in[5];
    float* out = (float*)d_out;

    dim3 blocks(OUT_DIM / TILE_O, BATCH / TILE_B);   // (2, 256)
    size_t wbytes = (size_t)OUT_DIM * KTOT * 2u;     // 6 MB f16 packed W
    if (ws_size >= wbytes) {
        _Float16* wpack = (_Float16*)d_ws;
        kan_pack_w<<<(OUT_DIM * KTOT + 255) / 256, 256, 0, stream>>>(cf, sb, ss, wpack);
        kan_gemm<true><<<blocks, NTHREADS, 0, stream>>>(x, g, cf, sb, ss, bs, wpack, out);
    } else {
        kan_gemm<false><<<blocks, NTHREADS, 0, stream>>>(x, g, cf, sb, ss, bs, nullptr, out);
    }
}
